// ODEModel_57664230916361
// MI455X (gfx1250) — matrix operand order repaired and checked by reading, not compiled, not run on hardware
//
#include <hip/hip_runtime.h>
#include <cmath>

#define NDIM   4
#define BOBJ   8
#define NSYS   1024
#define NB     (NSYS * BOBJ)               // 8192 object rows
#define NEDGE  (NSYS * BOBJ * (BOBJ - 1))  // 57344 edge rows
#define HF     256
#define HG     512
#define TSTEPS 3

#define FROWS  64   // rows per block, f-MLP  (LDS: 4 + 2*32 = 68 KB)
#define GROWS  64   // edges per block, g-MLP (LDS: 4 + 2*64 = 132 KB)

typedef __attribute__((ext_vector_type(16))) _Float16 v16h;
typedef __attribute__((ext_vector_type(8)))  _Float16 v8h;
typedef __attribute__((ext_vector_type(8)))  float    v8f;

// ---------------------------------------------------------------------------
// helpers
// ---------------------------------------------------------------------------
// branch-free softplus on raw HW transcendentals (arg of log is in [1,2],
// never subnormal, so skip libm's denorm guards):
//   softplus(x) = max(x,0) + ln2 * log2(1 + exp2(-|x|*log2e))
static __device__ __forceinline__ float softplusf(float x) {
  float t = __builtin_amdgcn_exp2f(-1.4426950408889634f * fabsf(x));
  return fmaxf(x, 0.0f) + 0.6931471805599453f * __builtin_amdgcn_logf(1.0f + t);
}

// A fragment (16x32 f16, M x K) from row-major f16 activations in LDS.
// ISA layout: lane(0..15): M=lane, VGPR0..3 K=0..7, VGPR4..7 K=16..23
//             lane(16..31): M=lane-16, K=8..15 / 24..31
static __device__ __forceinline__ v16h load_a_frag(const _Float16* __restrict__ base,
                                                   int stride, int kbase, int lane) {
  const int mrow = lane & 15;
  const int half = lane >> 4;
  const _Float16* p = base + mrow * stride + kbase + half * 8;
  v8h lo = *(const v8h*)(p);        // K = half*8 + 0..7
  v8h hi = *(const v8h*)(p + 16);   // K = 16 + half*8 + 0..7
  return __builtin_shufflevector(lo, hi, 0,1,2,3,4,5,6,7,8,9,10,11,12,13,14,15);
}

// B fragment: weights pre-repacked so each lane's 16 halves are contiguous.
static __device__ __forceinline__ v16h load_b_frag(const _Float16* __restrict__ frag,
                                                   int tileIdx, int lane) {
  return *(const v16h*)(frag + (size_t)tileIdx * 512 + lane * 16);
}

static __device__ __forceinline__ v8f wmma_f16(v16h a, v16h b, v8f c) {
  return __builtin_amdgcn_wmma_f32_16x16x32_f16(false, a, false, b, (short)0, c,
                                                false, false);
}

// ---------------------------------------------------------------------------
// weight repack: f32 (K x N) -> f16 fragment-native layout
// frag[(ct*ktiles + kt)*512 + lane*16 + e] = W[kt*32 + (e>>3)*16 + (lane>>4)*8 + (e&7)][ct*16 + (lane&15)]
// ---------------------------------------------------------------------------
__global__ void repack_kernel(const float* __restrict__ W, _Float16* __restrict__ frag,
                              int K, int N, int ktiles, int total) {
  int idx = blockIdx.x * blockDim.x + threadIdx.x;
  if (idx >= total) return;
  int e    = idx & 15;
  int lane = (idx >> 4) & 31;
  int tile = idx >> 9;
  int kt   = tile % ktiles;
  int ct   = tile / ktiles;
  int k = kt * 32 + ((e >> 3) << 4) + ((lane >> 4) << 3) + (e & 7);
  int n = ct * 16 + (lane & 15);
  float v = (k < K && n < N) ? W[(size_t)k * N + n] : 0.0f;
  frag[idx] = (_Float16)v;
}

// ---------------------------------------------------------------------------
// self MLP f: 8 -> 256 -> 256 -> 8, fused, FROWS rows per block, 8 waves.
// Each wave owns one row tile (rt = wave%4) and walks column tiles, so all
// A fragments for a layer are loaded from LDS once and reused from VGPRs.
// ---------------------------------------------------------------------------
__global__ __launch_bounds__(256) void f_mlp_kernel(
    const float* __restrict__ z,
    const _Float16* __restrict__ w0, const float* __restrict__ b0,
    const _Float16* __restrict__ w1, const float* __restrict__ b1,
    const _Float16* __restrict__ w2, const float* __restrict__ b2,
    float* __restrict__ dzself) {
  __shared__ __align__(32) _Float16 Ain[FROWS * 32];
  __shared__ __align__(32) _Float16 H1[FROWS * HF];
  __shared__ __align__(32) _Float16 H2[FROWS * HF];
  const int tid  = threadIdx.x;
  const int lane = tid & 31;
  const int wave = tid >> 5;
  const int r0   = blockIdx.x * FROWS;
  const int ncol = lane & 15;
  const int half = lane >> 4;
  const int RT   = FROWS / 16;       // 4 row tiles
  const int rt   = wave % RT;        // fixed row tile per wave
  const int c0   = wave / RT;        // 0..1 (2 waves share a row tile)

  // stage input rows (K padded 8 -> 32)
  for (int i = tid; i < FROWS * 32; i += 256) {
    int r = i >> 5, c = i & 31;
    Ain[i] = (_Float16)((c < 8) ? z[(size_t)(r0 + r) * 8 + c] : 0.0f);
  }
  __syncthreads();

  // layer 0: K=32 (single WMMA per tile), A fragment hoisted
  {
    v16h a0 = load_a_frag(Ain + rt * 16 * 32, 32, 0, lane);
    for (int ct = c0; ct < HF / 16; ct += 2) {
      v8f acc = {};
      acc = wmma_f16(a0, load_b_frag(w0, ct, lane), acc);
      float bv = b0[ct * 16 + ncol];
#pragma unroll
      for (int v = 0; v < 8; ++v)
        H1[(rt * 16 + half * 8 + v) * HF + ct * 16 + ncol] =
            (_Float16)softplusf(acc[v] + bv);
    }
  }
  __syncthreads();

  // layer 1: K=256 -> 8 WMMAs per tile, all A fragments hoisted (64 VGPRs)
  {
    v16h af[HF / 32];
#pragma unroll
    for (int kt = 0; kt < HF / 32; ++kt)
      af[kt] = load_a_frag(H1 + rt * 16 * HF, HF, kt * 32, lane);
    for (int ct = c0; ct < HF / 16; ct += 2) {
      v8f acc = {};
#pragma unroll
      for (int kt = 0; kt < HF / 32; ++kt)
        acc = wmma_f16(af[kt], load_b_frag(w1, ct * (HF / 32) + kt, lane), acc);
      float bv = b1[ct * 16 + ncol];
#pragma unroll
      for (int v = 0; v < 8; ++v)
        H2[(rt * 16 + half * 8 + v) * HF + ct * 16 + ncol] =
            (_Float16)softplusf(acc[v] + bv);
    }
  }
  __syncthreads();

  // layer 2: N=16 (8 real), K=256; waves with c0==0 handle their row tile
  if (c0 == 0) {
    v8f acc = {};
#pragma unroll
    for (int kt = 0; kt < HF / 32; ++kt)
      acc = wmma_f16(load_a_frag(H2 + rt * 16 * HF, HF, kt * 32, lane),
                     load_b_frag(w2, kt, lane), acc);
    if (ncol < 8) {
      float bv = b2[ncol];
#pragma unroll
      for (int v = 0; v < 8; ++v)
        dzself[(size_t)(r0 + rt * 16 + half * 8 + v) * 8 + ncol] = acc[v] + bv;
    }
  }
}

// ---------------------------------------------------------------------------
// interaction MLP g: 12 -> 512 -> 512 -> 8, fused, GROWS edges per block
// ---------------------------------------------------------------------------
__global__ __launch_bounds__(256) void g_mlp_kernel(
    const float* __restrict__ z,
    const int* __restrict__ rec_idx, const int* __restrict__ snd_idx,
    const _Float16* __restrict__ w0, const float* __restrict__ b0,
    const _Float16* __restrict__ w1, const float* __restrict__ b1,
    const _Float16* __restrict__ w2, const float* __restrict__ b2,
    float* __restrict__ dzint) {
  __shared__ __align__(32) _Float16 Ain[GROWS * 32];
  __shared__ __align__(32) _Float16 H1[GROWS * HG];
  __shared__ __align__(32) _Float16 H2[GROWS * HG];
  const int tid  = threadIdx.x;
  const int lane = tid & 31;
  const int wave = tid >> 5;
  const int e0   = blockIdx.x * GROWS;
  const int ncol = lane & 15;
  const int half = lane >> 4;
  const int RT   = GROWS / 16;       // 4 row tiles
  const int rt   = wave % RT;
  const int c0   = wave / RT;

  // build edge input: [recv_pos - send_pos, recv_vel, send_vel], pad 12 -> 32
  for (int i = tid; i < GROWS * 32; i += 256) {
    int r = i >> 5, c = i & 31;
    int edge = e0 + r;
    int sys  = edge / 56;
    int pair = edge % 56;
    size_t rrow = (size_t)(sys * 8 + rec_idx[pair]) * 8;
    size_t srow = (size_t)(sys * 8 + snd_idx[pair]) * 8;
    float v = 0.0f;
    if (c < 4)       v = z[rrow + c] - z[srow + c];
    else if (c < 8)  v = z[rrow + c];
    else if (c < 12) v = z[srow + (c - 4)];
    Ain[i] = (_Float16)v;
  }
  __syncthreads();

  // layer 0: K=32, 32 column tiles, A fragment hoisted
  {
    v16h a0 = load_a_frag(Ain + rt * 16 * 32, 32, 0, lane);
    for (int ct = c0; ct < HG / 16; ct += 2) {
      v8f acc = {};
      acc = wmma_f16(a0, load_b_frag(w0, ct, lane), acc);
      float bv = b0[ct * 16 + ncol];
#pragma unroll
      for (int v = 0; v < 8; ++v)
        H1[(rt * 16 + half * 8 + v) * HG + ct * 16 + ncol] =
            (_Float16)softplusf(acc[v] + bv);
    }
  }
  __syncthreads();

  // layer 1: K=512 -> 16 WMMAs per tile, all A fragments hoisted (128 VGPRs)
  {
    v16h af[HG / 32];
#pragma unroll
    for (int kt = 0; kt < HG / 32; ++kt)
      af[kt] = load_a_frag(H1 + rt * 16 * HG, HG, kt * 32, lane);
    for (int ct = c0; ct < HG / 16; ct += 2) {
      v8f acc = {};
#pragma unroll
      for (int kt = 0; kt < HG / 32; ++kt)
        acc = wmma_f16(af[kt], load_b_frag(w1, ct * (HG / 32) + kt, lane), acc);
      float bv = b1[ct * 16 + ncol];
#pragma unroll
      for (int v = 0; v < 8; ++v)
        H2[(rt * 16 + half * 8 + v) * HG + ct * 16 + ncol] =
            (_Float16)softplusf(acc[v] + bv);
    }
  }
  __syncthreads();

  // layer 2: RT tiles of 16x16 (8 real cols), K=512
  if (c0 == 0) {
    v8f acc = {};
#pragma unroll
    for (int kt = 0; kt < HG / 32; ++kt)
      acc = wmma_f16(load_a_frag(H2 + rt * 16 * HG, HG, kt * 32, lane),
                     load_b_frag(w2, kt, lane), acc);
    if (ncol < 8) {
      float bv = b2[ncol];
#pragma unroll
      for (int v = 0; v < 8; ++v)
        dzint[(size_t)(e0 + rt * 16 + half * 8 + v) * 8 + ncol] = acc[v] + bv;
    }
  }
}

// ---------------------------------------------------------------------------
// combine: dz = dz_self + sum_{j<7} dz_int[row*7+j];  znext = zbase + alpha*h*dz
// ---------------------------------------------------------------------------
__global__ __launch_bounds__(256) void combine_kernel(
    const float* __restrict__ zbase, const float* __restrict__ dzself,
    const float* __restrict__ dzint, float* __restrict__ kout,
    float* __restrict__ znext, const float* __restrict__ ts, float alpha) {
  int idx = blockIdx.x * blockDim.x + threadIdx.x;
  if (idx >= NB * 8) return;
  int row = idx >> 3, c = idx & 7;
  float s = dzself[idx];
  const float* p = dzint + (size_t)row * 7 * 8 + c;
#pragma unroll
  for (int j = 0; j < 7; ++j) s += p[j * 8];
  kout[idx] = s;
  float h = ts[1] - ts[0];
  znext[idx] = zbase[idx] + alpha * h * s;
}

// ---------------------------------------------------------------------------
// RK4 update + trajectory write
// ---------------------------------------------------------------------------
__global__ __launch_bounds__(256) void update_kernel(
    float* __restrict__ zcur, const float* __restrict__ k1,
    const float* __restrict__ k2, const float* __restrict__ k3,
    const float* __restrict__ k4, const float* __restrict__ ts,
    float* __restrict__ out, int tslot) {
  int idx = blockIdx.x * blockDim.x + threadIdx.x;
  if (idx >= NB * 8) return;
  float h  = ts[1] - ts[0];
  float zn = zcur[idx] +
             h * (1.0f / 6.0f) * (k1[idx] + 2.0f * k2[idx] + 2.0f * k3[idx] + k4[idx]);
  zcur[idx] = zn;
  int row = idx >> 3, c = idx & 7;
  out[(size_t)row * (TSTEPS * 8) + tslot * 8 + c] = zn;
}

__global__ __launch_bounds__(256) void init_kernel(const float* __restrict__ zd0,
                                                   float* __restrict__ zcur,
                                                   float* __restrict__ out) {
  int idx = blockIdx.x * blockDim.x + threadIdx.x;
  if (idx >= NB * 8) return;
  zcur[idx] = zd0[idx];
  int row = idx >> 3, c = idx & 7;
  out[(size_t)row * (TSTEPS * 8) + c] = zd0[idx];
}

// ---------------------------------------------------------------------------
// host
// ---------------------------------------------------------------------------
extern "C" void kernel_launch(void* const* d_in, const int* in_sizes, int n_in,
                              void* d_out, int out_size, void* d_ws, size_t ws_size,
                              hipStream_t stream) {
  const float* zd0 = (const float*)d_in[0];
  const float* ts  = (const float*)d_in[1];
  const float* fW0 = (const float*)d_in[2];
  const float* fb0 = (const float*)d_in[3];
  const float* fW1 = (const float*)d_in[4];
  const float* fb1 = (const float*)d_in[5];
  const float* fW2 = (const float*)d_in[6];
  const float* fb2 = (const float*)d_in[7];
  const float* gW0 = (const float*)d_in[8];
  const float* gb0 = (const float*)d_in[9];
  const float* gW1 = (const float*)d_in[10];
  const float* gb1 = (const float*)d_in[11];
  const float* gW2 = (const float*)d_in[12];
  const float* gb2 = (const float*)d_in[13];
  const int*   rec = (const int*)d_in[14];
  const int*   snd = (const int*)d_in[15];
  float* out = (float*)d_out;

  // workspace carve-out (256B aligned)
  char* w = (char*)d_ws;
  auto carve = [&](size_t bytes) {
    void* p = (void*)w;
    w += (bytes + 255) & ~(size_t)255;
    return p;
  };
  float* zcur   = (float*)carve((size_t)NB * 8 * 4);
  float* ztmp   = (float*)carve((size_t)NB * 8 * 4);
  float* k1     = (float*)carve((size_t)NB * 8 * 4);
  float* k2     = (float*)carve((size_t)NB * 8 * 4);
  float* k3     = (float*)carve((size_t)NB * 8 * 4);
  float* k4     = (float*)carve((size_t)NB * 8 * 4);
  float* dzself = (float*)carve((size_t)NB * 8 * 4);
  float* dzint  = (float*)carve((size_t)NEDGE * 8 * 4);
  _Float16* fw0 = (_Float16*)carve((size_t)16 * 1 * 512 * 2);
  _Float16* fw1 = (_Float16*)carve((size_t)16 * 8 * 512 * 2);
  _Float16* fw2 = (_Float16*)carve((size_t)1 * 8 * 512 * 2);
  _Float16* gw0 = (_Float16*)carve((size_t)32 * 1 * 512 * 2);
  _Float16* gw1 = (_Float16*)carve((size_t)32 * 16 * 512 * 2);
  _Float16* gw2 = (_Float16*)carve((size_t)1 * 16 * 512 * 2);

  auto repack = [&](const float* W, _Float16* frag, int K, int N, int ktiles,
                    int ctiles) {
    int total = ctiles * ktiles * 512;
    repack_kernel<<<(total + 255) / 256, 256, 0, stream>>>(W, frag, K, N, ktiles,
                                                           total);
  };
  repack(fW0, fw0, 8,   HF, 1,       HF / 16);
  repack(fW1, fw1, HF,  HF, HF / 32, HF / 16);
  repack(fW2, fw2, HF,  8,  HF / 32, 1);
  repack(gW0, gw0, 12,  HG, 1,       HG / 16);
  repack(gW1, gw1, HG,  HG, HG / 32, HG / 16);
  repack(gW2, gw2, HG,  8,  HG / 32, 1);

  const int NELEM = NB * 8;
  init_kernel<<<(NELEM + 255) / 256, 256, 0, stream>>>(zd0, zcur, out);

  auto rhs = [&](const float* zin, float* kout, float* znext, float alpha) {
    f_mlp_kernel<<<NB / FROWS, 256, 0, stream>>>(zin, fw0, fb0, fw1, fb1, fw2, fb2,
                                                 dzself);
    g_mlp_kernel<<<NEDGE / GROWS, 256, 0, stream>>>(zin, rec, snd, gw0, gb0, gw1,
                                                    gb1, gw2, gb2, dzint);
    combine_kernel<<<(NELEM + 255) / 256, 256, 0, stream>>>(zcur, dzself, dzint,
                                                            kout, znext, ts, alpha);
  };

  for (int step = 0; step < TSTEPS - 1; ++step) {
    rhs(zcur, k1, ztmp, 0.5f);
    rhs(ztmp, k2, ztmp, 0.5f);
    rhs(ztmp, k3, ztmp, 1.0f);
    rhs(ztmp, k4, ztmp, 0.0f);
    update_kernel<<<(NELEM + 255) / 256, 256, 0, stream>>>(zcur, k1, k2, k3, k4, ts,
                                                           out, step + 1);
  }
}
